// SelfAttention_81106162417714
// MI455X (gfx1250) — matrix operand order repaired and checked by reading, not compiled, not run on hardware
//
#include <hip/hip_runtime.h>

// ---------------------------------------------------------------------------
// Self-attention (B=2, S=4096, D=512, H=8, Hd=64) for gfx1250 / MI455X.
// All matmuls via v_wmma_f32_16x16x32_f16 (f16 inputs, f32 accumulate).
// Pipeline: [QKV gemm -> f16 head-major buffers] -> [flash attention] ->
//           [output projection -> fp32 d_out].
// v3: fix async builtin signature: (int4 AS1*, int4 AS3*, imm, imm).
// ---------------------------------------------------------------------------

typedef _Float16 h16;
typedef __attribute__((ext_vector_type(16))) _Float16 v16h;
typedef __attribute__((ext_vector_type(8)))  _Float16 v8h;
typedef __attribute__((ext_vector_type(8)))  float    v8f;
typedef __attribute__((ext_vector_type(4)))  float    v4f;
typedef __attribute__((ext_vector_type(4)))  int      v4i;

#define N_HEAD   8
#define N_EMBD   512
#define HEAD_DIM 64
#define BATCH    2
#define SEQ      4096
#define MROWS    (BATCH * SEQ)   // 8192

#define WMMA_F16(a, b, c) \
  __builtin_amdgcn_wmma_f32_16x16x32_f16(false, (a), false, (b), (short)0, (c), false, false)

// --- CDNA5 async global->LDS copy (ASYNCcnt path), with safe fallback -------
#if __has_builtin(__builtin_amdgcn_global_load_async_to_lds_b128) && \
    __has_builtin(__builtin_amdgcn_s_wait_asynccnt)
#define HAVE_ASYNC_LDS 1
typedef __attribute__((address_space(1))) v4i g_v4i;   // global int4
typedef __attribute__((address_space(3))) v4i l_v4i;   // LDS int4
// 16-byte async copy: global (gp) -> LDS (lp); tracked by ASYNCcnt.
static __device__ __forceinline__ void async_b128(const h16* gp, h16* lp) {
  __builtin_amdgcn_global_load_async_to_lds_b128(
      (g_v4i*)(void*)gp, (l_v4i*)(void*)lp, 0, 0);
}
static __device__ __forceinline__ void async_wait() {
  __builtin_amdgcn_s_wait_asynccnt(0);
}
#else
#define HAVE_ASYNC_LDS 0
static __device__ __forceinline__ void async_b128(const h16* gp, h16* lp) {
  *(v8h*)lp = *(const v8h*)gp;
}
static __device__ __forceinline__ void async_wait() {}
#endif

static __device__ __forceinline__ v16h cat8(v8h lo, v8h hi) {
  return __builtin_shufflevector(lo, hi, 0,1,2,3,4,5,6,7,8,9,10,11,12,13,14,15);
}

// A fragment (16x32 f16) from row-major [rows][ld] storage.
// ISA layout: lane l holds row M=l&15; element e -> K = (l>>4)*8 + (e&7) + (e>>3)*16.
static __device__ __forceinline__ v16h load_frag_a(const h16* p, int ld, int row0,
                                                   int kc, int lane) {
  int row = row0 + (lane & 15);
  int kb  = kc + ((lane >> 4) << 3);
  const h16* base = p + row * ld + kb;
  v8h lo = *(const v8h*)(base);
  v8h hi = *(const v8h*)(base + 16);
  return cat8(lo, hi);
}

// B fragment (32x16 f16) from TRANSPOSED storage bt[n][k] with stride ld.
// ISA layout: lane l holds column N=l&15; element e -> K = (l>>4)*8 + (e&7) + (e>>3)*16.
static __device__ __forceinline__ v16h load_frag_b(const h16* p, int ld, int n0,
                                                   int kc, int lane) {
  int n  = n0 + (lane & 15);
  int kb = kc + ((lane >> 4) << 3);
  const h16* base = p + n * ld + kb;
  v8h lo = *(const v8h*)(base);
  v8h hi = *(const v8h*)(base + 16);
  return cat8(lo, hi);
}

// ---------------------------------------------------------------------------
// Kernel 1: QKV projection. C = x @ W + b, stored f16 as [b][h][s][hd].
// grid = (N/64, M/128, 3), block = 128 (4 waves). Wave tile: 32x64
// (2 A-frags x 4 B-frags = 8 WMMAs per 32-deep K step).
// ---------------------------------------------------------------------------
__global__ __launch_bounds__(128)
void qkv_gemm_kernel(const float* __restrict__ x,
                     const float* __restrict__ Wq, const float* __restrict__ bq,
                     const float* __restrict__ Wk, const float* __restrict__ bk,
                     const float* __restrict__ Wv, const float* __restrict__ bv,
                     h16* __restrict__ qbuf, h16* __restrict__ kbuf,
                     h16* __restrict__ vbuf) {
  __shared__ h16 sA[128 * 32];  // A tile, row-major [m][k]
  __shared__ h16 sB[64 * 32];   // B tile, transposed [n][k]

  const int which = blockIdx.z;
  const float* W    = (which == 0) ? Wq : (which == 1) ? Wk : Wv;
  const float* bias = (which == 0) ? bq : (which == 1) ? bk : bv;
  h16* outb         = (which == 0) ? qbuf : (which == 1) ? kbuf : vbuf;

  const int n0 = blockIdx.x * 64;
  const int m0 = blockIdx.y * 128;
  const int tid  = threadIdx.x;
  const int lane = tid & 31;
  const int wave = tid >> 5;

  v8f acc0[4] = {};   // rows [wave*32, +16)
  v8f acc1[4] = {};   // rows [wave*32+16, +16)

  for (int k0 = 0; k0 < N_EMBD; k0 += 32) {
    // Stage A: 128x32 fp32 -> f16; each thread converts one full row segment.
    {
      const float* src = x + (size_t)(m0 + tid) * N_EMBD + k0;
      h16* dst = sA + tid * 32;
#pragma unroll
      for (int i = 0; i < 4; ++i) {
        v4f f0 = *(const v4f*)(src + i * 8);
        v4f f1 = *(const v4f*)(src + i * 8 + 4);
        v8h hv;
#pragma unroll
        for (int j = 0; j < 4; ++j) { hv[j] = (h16)f0[j]; hv[4 + j] = (h16)f1[j]; }
        *(v8h*)(dst + i * 8) = hv;
      }
      if (k0 + 32 < N_EMBD) __builtin_prefetch(src + 32, 0, 3);
    }
    // Stage B transposed: read W row-major (coalesced), scatter to [n][k]
    {
      int kr  = tid >> 2;          // k row 0..31
      int seg = (tid & 3) * 16;    // n segment
      const float* src = W + (size_t)(k0 + kr) * N_EMBD + n0 + seg;
#pragma unroll
      for (int i = 0; i < 4; ++i) {
        v4f f = *(const v4f*)(src + i * 4);
#pragma unroll
        for (int j = 0; j < 4; ++j) sB[(seg + i * 4 + j) * 32 + kr] = (h16)f[j];
      }
      if (k0 + 32 < N_EMBD) __builtin_prefetch(src + 32 * N_EMBD, 0, 3);
    }
    __syncthreads();

    v16h a0 = load_frag_a(sA, 32, wave * 32, 0, lane);
    v16h a1 = load_frag_a(sA, 32, wave * 32 + 16, 0, lane);
    v16h bf[4];
#pragma unroll
    for (int nb = 0; nb < 4; ++nb) bf[nb] = load_frag_b(sB, 32, nb * 16, 0, lane);
#pragma unroll
    for (int nb = 0; nb < 4; ++nb) acc0[nb] = WMMA_F16(a0, bf[nb], acc0[nb]);
#pragma unroll
    for (int nb = 0; nb < 4; ++nb) acc1[nb] = WMMA_F16(a1, bf[nb], acc1[nb]);
    __syncthreads();
  }

  // Epilogue: bias add, convert to f16, store head-major [b][h][s][hd]
#pragma unroll
  for (int nb = 0; nb < 4; ++nb) {
    int ncol = n0 + nb * 16 + (lane & 15);
    float bs = bias[ncol];
    int hh = ncol >> 6;
    int hd = ncol & 63;
#pragma unroll
    for (int blk = 0; blk < 2; ++blk) {
      const v8f* acc = blk ? acc1 : acc0;
#pragma unroll
      for (int v = 0; v < 8; ++v) {
        int m    = m0 + wave * 32 + blk * 16 + v + ((lane >> 4) << 3);
        int bidx = m >> 12;          // / SEQ
        int s    = m & (SEQ - 1);
        size_t addr = (((size_t)bidx * N_HEAD + hh) * SEQ + s) * HEAD_DIM + hd;
        outb[addr] = (h16)(acc[nb][v] + bs);
      }
    }
  }
}

// ---------------------------------------------------------------------------
// Kernel 2: flash attention over one (b, h, 64-query tile) per block.
// block = 128 (4 waves); each wave owns 16 query rows, full Hd=64 output.
// K chunk staged with async global->LDS; V staged transposed manually.
// ---------------------------------------------------------------------------
__global__ __launch_bounds__(128)
void attn_kernel(const h16* __restrict__ qbuf, const h16* __restrict__ kbuf,
                 const h16* __restrict__ vbuf, h16* __restrict__ attnout) {
  __shared__ h16   sK[64 * 64];        // K chunk, [key][hd] == Bt for Q@K^T
  __shared__ h16   sV[64 * 64];        // V chunk, transposed [hd][key] == Bt for P@V
  __shared__ float sS[4][16 * 64];     // per-wave raw scores
  __shared__ h16   sP[4][16 * 64];     // per-wave exp(scores) as f16 (A for P@V)
  __shared__ float sM[64], sL[64], sAlpha[64];

  const int q0 = blockIdx.x * 64;
  const int hh = blockIdx.y;
  const int bb = blockIdx.z;
  const int tid  = threadIdx.x;
  const int lane = tid & 31;
  const int wave = tid >> 5;

  const size_t head_base = ((size_t)bb * N_HEAD + hh) * (size_t)SEQ * HEAD_DIM;

  // Q fragments stay in registers for the whole kernel (2 frags cover K=64).
  v16h qfrag[2];
  {
    int row = q0 + wave * 16 + (lane & 15);
    int kb  = (lane >> 4) << 3;
    const h16* qp = qbuf + head_base + (size_t)row * HEAD_DIM;
#pragma unroll
    for (int kc = 0; kc < 2; ++kc) {
      v8h lo = *(const v8h*)(qp + kc * 32 + kb);
      v8h hi = *(const v8h*)(qp + kc * 32 + kb + 16);
      qfrag[kc] = cat8(lo, hi);
    }
  }

  if (tid < 64) { sM[tid] = -3.0e38f; sL[tid] = 0.0f; }

  v8f o[4] = {};

  for (int j0 = 0; j0 < SEQ; j0 += 64) {
    __syncthreads();   // protect sK/sV overwrite vs previous iteration readers
    // Stage K (async byte-copy) and V (manual transpose) for this key chunk.
    {
      int key = tid >> 1;
      int seg = (tid & 1) * 32;
      const h16* ksrc = kbuf + head_base + (size_t)(j0 + key) * HEAD_DIM + seg;
#pragma unroll
      for (int i = 0; i < 4; ++i)
        async_b128(ksrc + i * 8, sK + key * 64 + seg + i * 8);
      const h16* vsrc = vbuf + head_base + (size_t)(j0 + key) * HEAD_DIM + seg;
#pragma unroll
      for (int i = 0; i < 4; ++i) {
        v8h vv = *(const v8h*)(vsrc + i * 8);
#pragma unroll
        for (int j = 0; j < 8; ++j) sV[(seg + i * 8 + j) * 64 + key] = vv[j];
      }
      if (j0 + 64 < SEQ) {
        __builtin_prefetch(ksrc + 64 * HEAD_DIM, 0, 3);
        __builtin_prefetch(vsrc + 64 * HEAD_DIM, 0, 3);
      }
    }
    async_wait();
    __syncthreads();

    // S = Q @ K^T (raw), 8 WMMAs per wave, B-frags loaded in batches
    v8f sacc[4] = {};
#pragma unroll
    for (int kc = 0; kc < 2; ++kc) {
      v16h bf[4];
#pragma unroll
      for (int nb = 0; nb < 4; ++nb) bf[nb] = load_frag_b(sK, 64, nb * 16, kc * 32, lane);
#pragma unroll
      for (int nb = 0; nb < 4; ++nb) sacc[nb] = WMMA_F16(qfrag[kc], bf[nb], sacc[nb]);
    }
    // Spill scaled scores to per-wave LDS scratch
    const float scale = 0.125f;   // 1/sqrt(64)
#pragma unroll
    for (int nb = 0; nb < 4; ++nb) {
      int n = nb * 16 + (lane & 15);
#pragma unroll
      for (int v = 0; v < 8; ++v) {
        int r = v + ((lane >> 4) << 3);
        sS[wave][r * 64 + n] = sacc[nb][v] * scale;
      }
    }
    __syncthreads();

    // Online softmax: lanes 0..15 each own one of the wave's 16 rows.
    if (lane < 16) {
      int r  = lane;
      int gr = wave * 16 + r;
      const float* srow = &sS[wave][r * 64];
      float mx = -3.0e38f;
#pragma unroll
      for (int n = 0; n < 64; ++n) mx = fmaxf(mx, srow[n]);
      float mold  = sM[gr];
      float mnew  = fmaxf(mold, mx);
      float alpha = __expf(mold - mnew);
      float sum = 0.0f;
      h16* prow = &sP[wave][r * 64];
#pragma unroll
      for (int n = 0; n < 64; ++n) {
        float p = __expf(srow[n] - mnew);
        sum += p;
        prow[n] = (h16)p;
      }
      sM[gr] = mnew;
      sL[gr] = sL[gr] * alpha + sum;
      sAlpha[gr] = alpha;
    }
    __syncthreads();

    // Rescale running O accumulators by alpha(row)
#pragma unroll
    for (int v = 0; v < 8; ++v) {
      float a = sAlpha[wave * 16 + v + ((lane >> 4) << 3)];
#pragma unroll
      for (int nb = 0; nb < 4; ++nb) o[nb][v] *= a;
    }

    // O += P @ V, 8 WMMAs per wave
#pragma unroll
    for (int kc = 0; kc < 2; ++kc) {
      v16h pfrag = load_frag_a(&sP[wave][0], 64, 0, kc * 32, lane);
      v16h bf[4];
#pragma unroll
      for (int nb = 0; nb < 4; ++nb) bf[nb] = load_frag_b(sV, 64, nb * 16, kc * 32, lane);
#pragma unroll
      for (int nb = 0; nb < 4; ++nb) o[nb] = WMMA_F16(pfrag, bf[nb], o[nb]);
    }
  }

  // Finalize: divide by row sums, store f16 to [B*S][D] (head-interleaved cols)
  if (lane < 16) sAlpha[wave * 16 + lane] = 1.0f / sL[wave * 16 + lane];
  __syncthreads();
#pragma unroll
  for (int nb = 0; nb < 4; ++nb) {
    int hd = nb * 16 + (lane & 15);
#pragma unroll
    for (int v = 0; v < 8; ++v) {
      int rloc = wave * 16 + v + ((lane >> 4) << 3);
      float inv = sAlpha[rloc];
      size_t m = (size_t)bb * SEQ + q0 + rloc;
      attnout[m * N_EMBD + hh * HEAD_DIM + hd] = (h16)(o[nb][v] * inv);
    }
  }
}

// ---------------------------------------------------------------------------
// Kernel 3: output projection. out = attn @ Wo + bo (fp32 result).
// grid = (N/64, M/128), block = 128; wave tile 32x64. A tile staged async.
// ---------------------------------------------------------------------------
__global__ __launch_bounds__(128)
void out_gemm_kernel(const h16* __restrict__ attn, const float* __restrict__ Wo,
                     const float* __restrict__ bo, float* __restrict__ out) {
  __shared__ h16 sA[128 * 32];
  __shared__ h16 sB[64 * 32];

  const int n0 = blockIdx.x * 64;
  const int m0 = blockIdx.y * 128;
  const int tid  = threadIdx.x;
  const int lane = tid & 31;
  const int wave = tid >> 5;

  v8f acc0[4] = {};
  v8f acc1[4] = {};

  for (int k0 = 0; k0 < N_EMBD; k0 += 32) {
    // Stage A (already f16): async byte-copy, one row per thread (64B = 4xB128)
    {
      const h16* src = attn + (size_t)(m0 + tid) * N_EMBD + k0;
      h16* dst = sA + tid * 32;
#pragma unroll
      for (int i = 0; i < 4; ++i) async_b128(src + i * 8, dst + i * 8);
      if (k0 + 32 < N_EMBD) __builtin_prefetch(src + 32, 0, 3);
    }
    // Stage B transposed
    {
      int kr  = tid >> 2;
      int seg = (tid & 3) * 16;
      const float* src = Wo + (size_t)(k0 + kr) * N_EMBD + n0 + seg;
#pragma unroll
      for (int i = 0; i < 4; ++i) {
        v4f f = *(const v4f*)(src + i * 4);
#pragma unroll
        for (int j = 0; j < 4; ++j) sB[(seg + i * 4 + j) * 32 + kr] = (h16)f[j];
      }
      if (k0 + 32 < N_EMBD) __builtin_prefetch(src + 32 * N_EMBD, 0, 3);
    }
    async_wait();
    __syncthreads();

    v16h a0 = load_frag_a(sA, 32, wave * 32, 0, lane);
    v16h a1 = load_frag_a(sA, 32, wave * 32 + 16, 0, lane);
    v16h bf[4];
#pragma unroll
    for (int nb = 0; nb < 4; ++nb) bf[nb] = load_frag_b(sB, 32, nb * 16, 0, lane);
#pragma unroll
    for (int nb = 0; nb < 4; ++nb) acc0[nb] = WMMA_F16(a0, bf[nb], acc0[nb]);
#pragma unroll
    for (int nb = 0; nb < 4; ++nb) acc1[nb] = WMMA_F16(a1, bf[nb], acc1[nb]);
    __syncthreads();
  }

#pragma unroll
  for (int nb = 0; nb < 4; ++nb) {
    int ncol = n0 + nb * 16 + (lane & 15);
    float bs = bo[ncol];
#pragma unroll
    for (int blk = 0; blk < 2; ++blk) {
      const v8f* acc = blk ? acc1 : acc0;
#pragma unroll
      for (int v = 0; v < 8; ++v) {
        int m = m0 + wave * 32 + blk * 16 + v + ((lane >> 4) << 3);
        out[(size_t)m * N_EMBD + ncol] = acc[nb][v] + bs;
      }
    }
  }
}

// ---------------------------------------------------------------------------
// Host-side launch
// ---------------------------------------------------------------------------
extern "C" void kernel_launch(void* const* d_in, const int* in_sizes, int n_in,
                              void* d_out, int out_size, void* d_ws, size_t ws_size,
                              hipStream_t stream) {
  (void)in_sizes; (void)n_in; (void)out_size; (void)ws_size;
  const float* x  = (const float*)d_in[0];
  const float* Wq = (const float*)d_in[1];
  const float* bq = (const float*)d_in[2];
  const float* Wk = (const float*)d_in[3];
  const float* bk = (const float*)d_in[4];
  const float* Wv = (const float*)d_in[5];
  const float* bv = (const float*)d_in[6];
  const float* Wo = (const float*)d_in[7];
  const float* bo = (const float*)d_in[8];

  const size_t elems = (size_t)MROWS * N_EMBD;   // 8192*512
  h16* qbuf = (h16*)d_ws;
  h16* kbuf = qbuf + elems;
  h16* vbuf = kbuf + elems;
  h16* abuf = vbuf + elems;

  dim3 g1(N_EMBD / 64, MROWS / 128, 3);
  qkv_gemm_kernel<<<g1, 128, 0, stream>>>(x, Wq, bq, Wk, bk, Wv, bv,
                                          qbuf, kbuf, vbuf);

  dim3 g2(SEQ / 64, N_HEAD, BATCH);
  attn_kernel<<<g2, 128, 0, stream>>>(qbuf, kbuf, vbuf, abuf);

  dim3 g3(N_EMBD / 64, MROWS / 128, 1);
  out_gemm_kernel<<<g3, 128, 0, stream>>>(abuf, Wo, bo, (float*)d_out);
}
